// MyVQVAEDoubleAdapt_45157286150848
// MI455X (gfx1250) — matrix operand order repaired and checked by reading, not compiled, not run on hardware
//
#include <hip/hip_runtime.h>

// ---------- types ----------
typedef __bf16 bf16;
typedef __attribute__((ext_vector_type(16))) __bf16 v16bf;
typedef __attribute__((ext_vector_type(8)))  __bf16 v8bf;
typedef __attribute__((ext_vector_type(8)))  float  v8f;

#define BND __launch_bounds__(256)

// CDNA5 async global->LDS staging (inline asm). Set to 0 to fall back to
// synchronous global_load + ds_store staging with identical barriers.
#ifndef USE_ASYNC_LDS
#define USE_ASYNC_LDS 1
#endif

// ---------- model dims ----------
#define Bb 128
#define Cc 12
#define Tt 512
#define Pp 16
#define Dd 512
#define Hh 8
#define NSEQ (Bb*Cc)          // 1536
#define S_ENC 33
#define S_DEC 32
#define M_ENC (NSEQ*S_ENC)    // 50688
#define M_DEC (NSEQ*S_DEC)    // 49152
#define NC_ 256
#define NF_ 1024

__device__ __forceinline__ float gelu_f(float x) {
  float x3 = x * x * x;
  return 0.5f * x * (1.0f + tanhf(0.7978845608028654f * (x + 0.044715f * x3)));
}

__device__ __forceinline__ void epilogue_store(float v, float bv, int act,
                                               float* Cf, bf16* Cb, size_t off) {
  if (act == 3)      v = bv - 2.0f * v;
  else {
    v += bv;
    if (act == 1)      v = gelu_f(v);
    else if (act == 2) v = tanhf(v);
  }
  if (Cf) Cf[off] = v;
  if (Cb) Cb[off] = (bf16)v;
}

// =====================================================================
// GEMM (N % 128 == 0, M % 128 == 0, K % 64 == 0):
//   C[M,N] = act(A[M,K](bf16) * Bt[N,K](bf16)^T + bias)
// act: 0 none, 1 gelu, 2 tanh, 3 bias-2*val (VQ distance)
// Block = 8 waves; wave computes 16x128; block macro tile 128x128.
// B tile (128 cols x 64 k) staged in LDS (double-buffered, async DMA),
// shared by all 8 waves; 16 WMMAs per wave per staged slice.
// grid = (N/128, M/128)
// =====================================================================
#define KSTEP 64
#define LDS_STRIDE 144                  // bytes per column entry (128B data + pad)
#define LDS_BUF    (128 * LDS_STRIDE)   // 18432 B per buffer

__global__ BND void gemm_n128(const bf16* __restrict__ A,
                              const bf16* __restrict__ Bt,
                              float* __restrict__ Cf,
                              bf16*  __restrict__ Cb,
                              const float* __restrict__ bias,
                              int M, int N, int K, int act) {
  __shared__ __attribute__((aligned(32))) char lb[2 * LDS_BUF];
  const int tid  = threadIdx.x;
  const int wave = tid >> 5;
  const int lane = tid & 31;
  const int m0 = blockIdx.y * 128 + wave * 16;
  const int n0 = blockIdx.x * 128;

  const int cl  = lane & 15;
  const int row = m0 + cl;
  const int kg  = (lane >> 4) * 16;   // A K-subgroup base (0 or 16)
  const int rb2 = (lane >> 4) * 32;   // B byte offset within column (K rows 0/16)
  const bf16* arow = A + (size_t)row * K;

  // staging: 1024 x 16B units = 128 cols x 8 k-octets; 4 units per thread
  const int uq = tid & 7;             // 16B unit (1 of 8) within the 128B column entry
  const int c0 = tid >> 3;            // cols c0, c0+32, c0+64, c0+96
  const bf16* g0 = Bt + (size_t)(n0 + c0)      * K + uq * 8;
  const bf16* g1 = Bt + (size_t)(n0 + c0 + 32) * K + uq * 8;
  const bf16* g2 = Bt + (size_t)(n0 + c0 + 64) * K + uq * 8;
  const bf16* g3 = Bt + (size_t)(n0 + c0 + 96) * K + uq * 8;
  const int d0 = c0 * LDS_STRIDE + uq * 16;
#if USE_ASYNC_LDS
  const unsigned lds_base = (unsigned)(uintptr_t)(&lb[0]);
#endif

  v8f acc[8];
  #pragma unroll
  for (int t = 0; t < 8; ++t)
    #pragma unroll
    for (int i = 0; i < 8; ++i) acc[t][i] = 0.f;

  // ---- stage k-slice kk (64 deep) into buffer buf ----
  auto stage = [&](int kk, int buf) {
#if USE_ASYNC_LDS
    unsigned l0 = lds_base + (unsigned)(buf * LDS_BUF + d0);
    asm volatile("global_load_async_to_lds_b128 %0, %1, off"
                 :: "v"(l0), "v"(g0 + kk) : "memory");
    asm volatile("global_load_async_to_lds_b128 %0, %1, off"
                 :: "v"(l0 + 32u * LDS_STRIDE), "v"(g1 + kk) : "memory");
    asm volatile("global_load_async_to_lds_b128 %0, %1, off"
                 :: "v"(l0 + 64u * LDS_STRIDE), "v"(g2 + kk) : "memory");
    asm volatile("global_load_async_to_lds_b128 %0, %1, off"
                 :: "v"(l0 + 96u * LDS_STRIDE), "v"(g3 + kk) : "memory");
#else
    v8bf t0 = *(const v8bf*)(g0 + kk);
    v8bf t1 = *(const v8bf*)(g1 + kk);
    v8bf t2 = *(const v8bf*)(g2 + kk);
    v8bf t3 = *(const v8bf*)(g3 + kk);
    *(v8bf*)(&lb[buf * LDS_BUF + d0])                   = t0;
    *(v8bf*)(&lb[buf * LDS_BUF + d0 +  32 * LDS_STRIDE]) = t1;
    *(v8bf*)(&lb[buf * LDS_BUF + d0 +  64 * LDS_STRIDE]) = t2;
    *(v8bf*)(&lb[buf * LDS_BUF + d0 +  96 * LDS_STRIDE]) = t3;
#endif
  };
  auto stage_wait = [&]() {
#if USE_ASYNC_LDS
    asm volatile("s_wait_asynccnt 0x0" ::: "memory");
#endif
    __syncthreads();
  };

  stage(0, 0);
  stage_wait();
  int cur = 0;
  union Au { v16bf v; v8bf h[2]; };
  for (int k = 0; k < K; k += KSTEP) {
    const bool more = (k + KSTEP) < K;
    if (more) stage(k + KSTEP, cur ^ 1);

    Au au0, au1;
    au0.h[0] = *(const v8bf*)(arow + k + kg);
    au0.h[1] = *(const v8bf*)(arow + k + kg + 8);
    au1.h[0] = *(const v8bf*)(arow + k + 32 + kg);
    au1.h[1] = *(const v8bf*)(arow + k + 32 + kg + 8);
    const char* bufp = &lb[cur * LDS_BUF];

    Au bu[8];
    // slice 0: k .. k+31  (bytes 0..63 of each column entry)
    #pragma unroll
    for (int ft = 0; ft < 8; ++ft) {
      const char* cp = bufp + (ft * 16 + cl) * LDS_STRIDE + rb2;
      bu[ft].h[0] = *(const v8bf*)(cp);
      bu[ft].h[1] = *(const v8bf*)(cp + 16);
    }
    #pragma unroll
    for (int ft = 0; ft < 8; ++ft)
      acc[ft] = __builtin_amdgcn_wmma_f32_16x16x32_bf16(
          false, au0.v, false, bu[ft].v, (short)0, acc[ft], false, false);
    // slice 1: k+32 .. k+63 (bytes 64..127 of each column entry)
    #pragma unroll
    for (int ft = 0; ft < 8; ++ft) {
      const char* cp = bufp + (ft * 16 + cl) * LDS_STRIDE + 64 + rb2;
      bu[ft].h[0] = *(const v8bf*)(cp);
      bu[ft].h[1] = *(const v8bf*)(cp + 16);
    }
    #pragma unroll
    for (int ft = 0; ft < 8; ++ft)
      acc[ft] = __builtin_amdgcn_wmma_f32_16x16x32_bf16(
          false, au1.v, false, bu[ft].v, (short)0, acc[ft], false, false);

    if (more) stage_wait();
    cur ^= 1;
  }

  const int rbase = m0 + 8 * (lane >> 4);
  #pragma unroll
  for (int ft = 0; ft < 8; ++ft) {
    const int col = n0 + ft * 16 + cl;
    const float bv = bias ? bias[col] : 0.f;
    #pragma unroll
    for (int i = 0; i < 8; ++i)
      epilogue_store(acc[ft][i], bv, act, Cf, Cb, (size_t)(rbase + i) * N + col);
  }
}

// ---- GEMM for N == 16 (head Wr2): one 16x16 tile per wave ----
__global__ BND void gemm_n16(const bf16* __restrict__ A,
                             const bf16* __restrict__ Bt,
                             float* __restrict__ Cf,
                             const float* __restrict__ bias,
                             int M, int K) {
  const int wave = threadIdx.x >> 5;
  const int lane = threadIdx.x & 31;
  const int m0 = blockIdx.y * 128 + wave * 16;
  const int row = m0 + (lane & 15);
  const int kg  = (lane >> 4) * 16;
  const int rb  = (lane >> 4) * 16;

  v8f acc;
  #pragma unroll
  for (int i = 0; i < 8; ++i) acc[i] = 0.f;
  const bf16* arow = A + (size_t)row * K;
  const bf16* b0 = Bt + (size_t)(lane & 15) * K + rb;
  for (int k = 0; k < K; k += 32) {
    union { v16bf v; v8bf h[2]; } au;
    au.h[0] = *(const v8bf*)(arow + k + kg);
    au.h[1] = *(const v8bf*)(arow + k + kg + 8);
    v16bf bb = *(const v16bf*)(b0 + k);
    acc = __builtin_amdgcn_wmma_f32_16x16x32_bf16(false, au.v, false, bb, (short)0, acc, false, false);
  }
  const int rbase = m0 + 8 * (lane >> 4);
  const int cl = lane & 15;
  const float bv = bias ? bias[cl] : 0.f;
  #pragma unroll
  for (int i = 0; i < 8; ++i)
    Cf[(size_t)(rbase + i) * 16 + cl] = acc[i] + bv;
}

// ---------- weight convert / transpose ----------
__global__ BND void cvt_transpose(const float* __restrict__ W, bf16* __restrict__ Wt,
                                  int K, int N) {
  long i = (long)blockIdx.x * 256 + threadIdx.x;
  if (i >= (long)K * N) return;
  int k = (int)(i / N), n = (int)(i % N);
  Wt[(size_t)n * K + k] = (bf16)W[i];
}
__global__ BND void cvt_bf16(const float* __restrict__ W, bf16* __restrict__ O, long n) {
  long i = (long)blockIdx.x * 256 + threadIdx.x;
  if (i < n) O[i] = (bf16)W[i];
}
__global__ BND void cbnorm_kernel(const float* __restrict__ cb, float* __restrict__ nrm,
                                  int ncodes) {
  int c = blockIdx.x * 256 + threadIdx.x;
  if (c >= ncodes) return;
  float s = 0.f;
  const float* p = cb + (size_t)c * Dd;
  for (int d = 0; d < Dd; ++d) { float v = p[d]; s += v * v; }
  nrm[c] = s;
}

// ---------- patch embed: gelu(xp@Wemb+bemb) -> LN(pn) -> +posenc ----------
__global__ BND void embed_kernel(const float* __restrict__ x, const float* __restrict__ Wemb,
                                 const float* __restrict__ bemb, const float* __restrict__ pn,
                                 float* __restrict__ h, bf16* __restrict__ hb) {
  const int tok = blockIdx.x;                // [0, NSEQ*32)
  const int tp = tok & 31, seq = tok >> 5;
  __shared__ float xs[16];
  __shared__ float vals[512];
  __shared__ float red[256];
  __shared__ float mv[2];
  const int t = threadIdx.x;
  if (t < 16) xs[t] = x[(size_t)seq * Tt + tp * 16 + t];
  __syncthreads();
  #pragma unroll
  for (int e = 0; e < 2; ++e) {
    int d = t + e * 256;
    float v = bemb[d];
    #pragma unroll
    for (int p = 0; p < 16; ++p) v += xs[p] * Wemb[p * Dd + d];
    vals[d] = gelu_f(v);
  }
  __syncthreads();
  red[t] = vals[t] + vals[t + 256];
  __syncthreads();
  for (int o = 128; o > 0; o >>= 1) { if (t < o) red[t] += red[t + o]; __syncthreads(); }
  if (!t) mv[0] = red[0] * (1.0f / 512.0f);
  __syncthreads();
  float m = mv[0];
  float d0 = vals[t] - m, d1 = vals[t + 256] - m;
  red[t] = d0 * d0 + d1 * d1;
  __syncthreads();
  for (int o = 128; o > 0; o >>= 1) { if (t < o) red[t] += red[t + o]; __syncthreads(); }
  if (!t) mv[1] = red[0] * (1.0f / 512.0f);
  __syncthreads();
  float inv = rsqrtf(mv[1] + 1e-5f);
  const int spos = tp + 1;
  #pragma unroll
  for (int e = 0; e < 2; ++e) {
    int d = t + e * 256;
    float xn = pn[d] * (vals[d] - m) * inv + pn[Dd + d];
    float f = expf(-(float)(d & ~1) * (9.210340371976184f / 512.0f));
    xn += (d & 1) ? cosf(spos * f) : sinf(spos * f);
    size_t off = ((size_t)seq * S_ENC + spos) * Dd + d;
    h[off] = xn; hb[off] = (bf16)xn;
  }
}

__global__ BND void cls_kernel(const float* __restrict__ cls, float* __restrict__ h,
                               bf16* __restrict__ hb) {
  int i = blockIdx.x * 256 + threadIdx.x;
  if (i >= NSEQ * Dd) return;
  int seq = i / Dd, d = i % Dd;
  float v = cls[d] + ((d & 1) ? 1.0f : 0.0f);   // posenc @ s=0: sin(0)=0, cos(0)=1
  size_t off = ((size_t)seq * S_ENC) * Dd + d;
  h[off] = v; hb[off] = (bf16)v;
}

// ---------- attention (scalar; small S) ----------
template <int S>
__global__ BND void attention_kernel(const float* __restrict__ qkv, bf16* __restrict__ ao,
                                     int nseq) {
  int idx = blockIdx.x * 256 + threadIdx.x;
  if (idx >= nseq * Hh * S) return;
  int s = idx % S, hd = (idx / S) % Hh, seq = idx / (S * Hh);
  size_t base = (size_t)seq * S;
  const float* qp = qkv + (base + s) * 1536 + hd * 64;
  float q[64];
  #pragma unroll
  for (int d = 0; d < 64; ++d) q[d] = qp[d];
  float sc[S]; float mx = -1e30f;
  #pragma unroll 1
  for (int tt = 0; tt < S; ++tt) {
    const float* kp = qkv + (base + tt) * 1536 + 512 + hd * 64;
    float dot = 0.f;
    #pragma unroll
    for (int d = 0; d < 64; ++d) dot += q[d] * kp[d];
    dot *= 0.125f;
    sc[tt] = dot; mx = fmaxf(mx, dot);
  }
  float den = 0.f;
  #pragma unroll 1
  for (int tt = 0; tt < S; ++tt) { sc[tt] = expf(sc[tt] - mx); den += sc[tt]; }
  float invd = 1.0f / den;
  float o[64];
  #pragma unroll
  for (int d = 0; d < 64; ++d) o[d] = 0.f;
  #pragma unroll 1
  for (int tt = 0; tt < S; ++tt) {
    float p = sc[tt] * invd;
    const float* vp = qkv + (base + tt) * 1536 + 1024 + hd * 64;
    #pragma unroll
    for (int d = 0; d < 64; ++d) o[d] += p * vp[d];
  }
  bf16* op = ao + (base + s) * Dd + hd * 64;
  #pragma unroll
  for (int d = 0; d < 64; ++d) op[d] = (bf16)o[d];
}

// ---------- residual + layernorm (row = 512) ----------
__global__ BND void add_ln_kernel(float* __restrict__ h, const float* __restrict__ delta,
                                  const float* __restrict__ gamma, const float* __restrict__ beta,
                                  bf16* __restrict__ hb) {
  const int row = blockIdx.x;
  const size_t base = (size_t)row * Dd;
  __shared__ float red[256];
  __shared__ float mv[2];
  const int t = threadIdx.x;
  float a = h[base + t] + delta[base + t];
  float b = h[base + t + 256] + delta[base + t + 256];
  red[t] = a + b;
  __syncthreads();
  for (int o = 128; o > 0; o >>= 1) { if (t < o) red[t] += red[t + o]; __syncthreads(); }
  if (!t) mv[0] = red[0] * (1.0f / 512.0f);
  __syncthreads();
  float m = mv[0];
  float da = a - m, db = b - m;
  red[t] = da * da + db * db;
  __syncthreads();
  for (int o = 128; o > 0; o >>= 1) { if (t < o) red[t] += red[t + o]; __syncthreads(); }
  if (!t) mv[1] = red[0] * (1.0f / 512.0f);
  __syncthreads();
  float inv = rsqrtf(mv[1] + 1e-5f);
  float ya = gamma[t]       * da * inv + beta[t];
  float yb = gamma[t + 256] * db * inv + beta[t + 256];
  h[base + t] = ya;        h[base + t + 256] = yb;
  hb[base + t] = (bf16)ya; hb[base + t + 256] = (bf16)yb;
}

// ---------- channel mix: out[b,o,s,:] = h[b,o,s,:] + sum_c h[b,c,s,:]*Wc[o,c] ----------
__global__ BND void chanmix_kernel(const float* __restrict__ h, const float* __restrict__ Wc,
                                   float* __restrict__ h2, bf16* __restrict__ hb,
                                   int S, int total) {
  int idx = blockIdx.x * 256 + threadIdx.x;
  if (idx >= total) return;
  int d = idx & 511;
  int rest = idx >> 9;
  int s = rest % S; rest /= S;
  int o = rest % Cc;
  int b = rest / Cc;
  size_t bbase = (size_t)b * Cc * S * Dd;
  float acc = h[idx];
  #pragma unroll
  for (int c = 0; c < Cc; ++c)
    acc += h[bbase + ((size_t)c * S + s) * Dd + d] * Wc[o * Cc + c];
  h2[idx] = acc; hb[idx] = (bf16)acc;
}

// ---------- VQ helpers ----------
__global__ BND void strip_cls_kernel(const float* __restrict__ h, float* __restrict__ ep,
                                     bf16* __restrict__ epb) {
  int idx = blockIdx.x * 256 + threadIdx.x;
  if (idx >= M_DEC * Dd) return;
  int d = idx & 511, row = idx >> 9;
  int s = row & 31, seq = row >> 5;
  float v = h[((size_t)seq * S_ENC + s + 1) * Dd + d];
  ep[idx] = v; epb[idx] = (bf16)v;
}

__global__ BND void argmin_gather_kernel(const float* __restrict__ scores, int ncodes,
                                         const float* __restrict__ cb, float* __restrict__ out) {
  const int row = blockIdx.x;
  __shared__ float bv[256];
  __shared__ int   bi[256];
  const int t = threadIdx.x;
  float best = 3.4e38f; int bidx = 0;
  for (int c = t; c < ncodes; c += 256) {
    float v = scores[(size_t)row * ncodes + c];
    if (v < best) { best = v; bidx = c; }
  }
  bv[t] = best; bi[t] = bidx;
  __syncthreads();
  for (int o = 128; o > 0; o >>= 1) {
    if (t < o) {
      float v2 = bv[t + o]; int i2 = bi[t + o];
      if (v2 < bv[t] || (v2 == bv[t] && i2 < bi[t])) { bv[t] = v2; bi[t] = i2; }
    }
    __syncthreads();
  }
  const int idx = bi[0];
  for (int d = t; d < Dd; d += 256)
    out[(size_t)row * Dd + d] = cb[(size_t)idx * Dd + d];
}

__global__ BND void sub_kernel(const float* __restrict__ a, const float* __restrict__ b,
                               float* __restrict__ r, bf16* __restrict__ rb, long n) {
  long i = (long)blockIdx.x * 256 + threadIdx.x;
  if (i >= n) return;
  float v = a[i] - b[i];
  r[i] = v; rb[i] = (bf16)v;
}
__global__ BND void add2_kernel(const float* __restrict__ a, const float* __restrict__ b,
                                float* __restrict__ r, bf16* __restrict__ rb, long n) {
  long i = (long)blockIdx.x * 256 + threadIdx.x;
  if (i >= n) return;
  float v = a[i] + b[i];
  r[i] = v; rb[i] = (bf16)v;
}

// ---------- reductions / output ----------
__global__ BND void zero_kernel(float* p, int n) {
  int i = blockIdx.x * 256 + threadIdx.x;
  if (i < n) p[i] = 0.f;
}
__global__ BND void mse_kernel(const float* __restrict__ a, const float* __restrict__ b,
                               float* __restrict__ acc, long n) {
  __shared__ float red[256];
  float s = 0.f;
  for (long i = (long)blockIdx.x * 256 + threadIdx.x; i < n; i += (long)gridDim.x * 256) {
    float d = a[i] - b[i]; s += d * d;
  }
  red[threadIdx.x] = s;
  __syncthreads();
  for (int o = 128; o > 0; o >>= 1) {
    if (threadIdx.x < o) red[threadIdx.x] += red[threadIdx.x + o];
    __syncthreads();
  }
  if (!threadIdx.x) atomicAdd(acc, red[0]);
}
__global__ void finalize_kernel(const float* __restrict__ acc, float* __restrict__ out) {
  out[0] = acc[0] / 786432.0f
         + 1.25f * (acc[1] / 25165824.0f)
         + 1.25f * (acc[2] / 25165824.0f);
}
__global__ BND void cls_rep_kernel(const float* __restrict__ h, float* __restrict__ out) {
  int i = blockIdx.x * 256 + threadIdx.x;
  if (i >= NSEQ * Dd) return;
  int seq = i / Dd, d = i % Dd;
  out[i] = h[((size_t)seq * S_ENC) * Dd + d];
}

// =====================================================================
// host orchestration
// =====================================================================
static inline int cdiv(long a, long b) { return (int)((a + b - 1) / b); }

extern "C" void kernel_launch(void* const* d_in, const int* in_sizes, int n_in,
                              void* d_out, int out_size, void* d_ws, size_t ws_size,
                              hipStream_t stream) {
  const float* x        = (const float*)d_in[0];
  const float* cls_tok  = (const float*)d_in[1];
  const float* Wemb     = (const float*)d_in[2];
  const float* bemb     = (const float*)d_in[3];
  const float* pn       = (const float*)d_in[4];
  const float* eWqkv    = (const float*)d_in[5];
  const float* eWo      = (const float*)d_in[6];
  const float* eLn1     = (const float*)d_in[7];
  const float* eWf1     = (const float*)d_in[8];
  const float* eWf2     = (const float*)d_in[9];
  const float* eLn2     = (const float*)d_in[10];
  const float* eWc      = (const float*)d_in[11];
  const float* dWqkv    = (const float*)d_in[12];
  const float* dWo      = (const float*)d_in[13];
  const float* dLn1     = (const float*)d_in[14];
  const float* dWf1     = (const float*)d_in[15];
  const float* dWf2     = (const float*)d_in[16];
  const float* dLn2     = (const float*)d_in[17];
  const float* dWc      = (const float*)d_in[18];
  const float* cbC      = (const float*)d_in[19];
  const float* cbF      = (const float*)d_in[20];
  const float* Wr1      = (const float*)d_in[21];
  const float* br1      = (const float*)d_in[22];
  const float* Wr2      = (const float*)d_in[23];
  const float* br2      = (const float*)d_in[24];
  float* fo = (float*)d_out;

  // ---- workspace carve ----
  char* base = (char*)d_ws;
  size_t off = 0;
  auto alloc = [&](size_t bytes) -> void* {
    void* p = base + off;
    off = (off + bytes + 255) & ~(size_t)255;
    return p;
  };
  float* hA  = (float*)alloc((size_t)M_ENC * Dd * 4);
  float* hB  = (float*)alloc((size_t)M_ENC * Dd * 4);
  bf16*  hb  = (bf16*) alloc((size_t)M_ENC * Dd * 2);
  float* qkv = (float*)alloc((size_t)M_ENC * 1536 * 4);
  bf16*  ao  = (bf16*) alloc((size_t)M_ENC * Dd * 2);
  bf16*  ffb = (bf16*) alloc((size_t)M_ENC * 2048 * 2);
  float* tmp = (float*)alloc((size_t)M_ENC * Dd * 4);
  float* ep  = (float*)alloc((size_t)M_DEC * Dd * 4);
  bf16*  epb = (bf16*) alloc((size_t)M_DEC * Dd * 2);
  float* cq  = (float*)alloc((size_t)M_DEC * Dd * 4);
  bf16*  rb  = (bf16*) alloc((size_t)M_DEC * Dd * 2);
  // aliases (phase-disjoint)
  float* scores = qkv;          // up to M_DEC x 1024 fp32
  float* rr     = tmp;          // fine residual
  float* fq     = (float*)ffb;  // fine quantized
  bf16*  t1b    = ao;           // head hidden

  bf16 *ewqkvT[3], *ewoT[3], *ewf1T[3], *ewf2T[3];
  bf16 *dwqkvT[3], *dwoT[3], *dwf1T[3], *dwf2T[3];
  for (int l = 0; l < 3; ++l) {
    ewqkvT[l] = (bf16*)alloc((size_t)1536 * 512 * 2);
    ewoT[l]   = (bf16*)alloc((size_t)512 * 512 * 2);
    ewf1T[l]  = (bf16*)alloc((size_t)2048 * 512 * 2);
    ewf2T[l]  = (bf16*)alloc((size_t)512 * 2048 * 2);
  }
  for (int l = 0; l < 3; ++l) {
    dwqkvT[l] = (bf16*)alloc((size_t)1536 * 512 * 2);
    dwoT[l]   = (bf16*)alloc((size_t)512 * 512 * 2);
    dwf1T[l]  = (bf16*)alloc((size_t)2048 * 512 * 2);
    dwf2T[l]  = (bf16*)alloc((size_t)512 * 2048 * 2);
  }
  bf16* wr1T = (bf16*)alloc((size_t)512 * 512 * 2);
  bf16* wr2T = (bf16*)alloc((size_t)16 * 512 * 2);
  bf16* cbCb = (bf16*)alloc((size_t)NC_ * Dd * 2);
  bf16* cbFb = (bf16*)alloc((size_t)NF_ * Dd * 2);
  float* cbnC = (float*)alloc(NC_ * 4);
  float* cbnF = (float*)alloc(NF_ * 4);
  float* accs = (float*)alloc(4 * 4);

  // ---- weight conversion ----
  auto tconv = [&](const float* W, bf16* Wt, int K, int N) {
    long n = (long)K * N;
    cvt_transpose<<<cdiv(n, 256), 256, 0, stream>>>(W, Wt, K, N);
  };
  for (int l = 0; l < 3; ++l) {
    tconv(eWqkv + (size_t)l * 512 * 1536, ewqkvT[l], 512, 1536);
    tconv(eWo   + (size_t)l * 512 * 512,  ewoT[l],   512, 512);
    tconv(eWf1  + (size_t)l * 512 * 2048, ewf1T[l],  512, 2048);
    tconv(eWf2  + (size_t)l * 2048 * 512, ewf2T[l],  2048, 512);
    tconv(dWqkv + (size_t)l * 512 * 1536, dwqkvT[l], 512, 1536);
    tconv(dWo   + (size_t)l * 512 * 512,  dwoT[l],   512, 512);
    tconv(dWf1  + (size_t)l * 512 * 2048, dwf1T[l],  512, 2048);
    tconv(dWf2  + (size_t)l * 2048 * 512, dwf2T[l],  2048, 512);
  }
  tconv(Wr1, wr1T, 512, 512);
  tconv(Wr2, wr2T, 512, 16);
  cvt_bf16<<<cdiv((long)NC_ * Dd, 256), 256, 0, stream>>>(cbC, cbCb, (long)NC_ * Dd);
  cvt_bf16<<<cdiv((long)NF_ * Dd, 256), 256, 0, stream>>>(cbF, cbFb, (long)NF_ * Dd);
  cbnorm_kernel<<<cdiv(NC_, 256), 256, 0, stream>>>(cbC, cbnC, NC_);
  cbnorm_kernel<<<cdiv(NF_, 256), 256, 0, stream>>>(cbF, cbnF, NF_);

  auto gemm = [&](const bf16* A, const bf16* Bt, float* Cf, bf16* Cb,
                  const float* bias, int M, int N, int K, int act) {
    dim3 grid(N / 128, M / 128);
    gemm_n128<<<grid, 256, 0, stream>>>(A, Bt, Cf, Cb, bias, M, N, K, act);
  };

  auto run_layer = [&](float*& hcur, float*& halt, int M, int S, int nseq,
                       const bf16* wq, const bf16* wo, const bf16* wf1, const bf16* wf2,
                       const float* ln1, const float* ln2, const float* Wc) {
    gemm(hb, wq, qkv, nullptr, nullptr, M, 1536, 512, 0);
    int at = nseq * Hh * S;
    if (S == S_ENC)
      attention_kernel<S_ENC><<<cdiv(at, 256), 256, 0, stream>>>(qkv, ao, nseq);
    else
      attention_kernel<S_DEC><<<cdiv(at, 256), 256, 0, stream>>>(qkv, ao, nseq);
    gemm(ao, wo, tmp, nullptr, nullptr, M, 512, 512, 0);
    add_ln_kernel<<<M, 256, 0, stream>>>(hcur, tmp, ln1, ln1 + 512, hb);
    gemm(hb, wf1, nullptr, ffb, nullptr, M, 2048, 512, 1);
    gemm(ffb, wf2, tmp, nullptr, nullptr, M, 512, 2048, 0);
    add_ln_kernel<<<M, 256, 0, stream>>>(hcur, tmp, ln2, ln2 + 512, hb);
    chanmix_kernel<<<cdiv((long)M * Dd, 256), 256, 0, stream>>>(hcur, Wc, halt, hb, S, M * Dd);
    float* t = hcur; hcur = halt; halt = t;
  };

  // ---- encoder ----
  embed_kernel<<<NSEQ * 32, 256, 0, stream>>>(x, Wemb, bemb, pn, hA, hb);
  cls_kernel<<<cdiv(NSEQ * Dd, 256), 256, 0, stream>>>(cls_tok, hA, hb);
  float *hcur = hA, *halt = hB;
  for (int l = 0; l < 3; ++l)
    run_layer(hcur, halt, M_ENC, S_ENC, NSEQ,
              ewqkvT[l], ewoT[l], ewf1T[l], ewf2T[l],
              eLn1 + (size_t)l * 1024, eLn2 + (size_t)l * 1024,
              eWc + (size_t)l * Cc * Cc);

  // ---- cls_rep out, strip cls ----
  cls_rep_kernel<<<cdiv(NSEQ * Dd, 256), 256, 0, stream>>>(hcur, fo + 786433);
  strip_cls_kernel<<<cdiv((long)M_DEC * Dd, 256), 256, 0, stream>>>(hcur, ep, epb);

  // ---- VQ coarse ----
  gemm(epb, cbCb, scores, nullptr, cbnC, M_DEC, NC_, 512, 3);
  argmin_gather_kernel<<<M_DEC, 256, 0, stream>>>(scores, NC_, cbC, cq);
  sub_kernel<<<cdiv((long)M_DEC * Dd, 256), 256, 0, stream>>>(ep, cq, rr, rb, (long)M_DEC * Dd);
  // ---- VQ fine ----
  gemm(rb, cbFb, scores, nullptr, cbnF, M_DEC, NF_, 512, 3);
  argmin_gather_kernel<<<M_DEC, 256, 0, stream>>>(scores, NF_, cbF, fq);

  // ---- VQ losses (before decoder clobbers aliases) ----
  zero_kernel<<<1, 32, 0, stream>>>(accs, 4);
  mse_kernel<<<1024, 256, 0, stream>>>(cq, ep, accs + 1, (long)M_DEC * Dd);
  mse_kernel<<<1024, 256, 0, stream>>>(fq, rr, accs + 2, (long)M_DEC * Dd);

  // ---- decoder input q = cq + fq ----
  float* hdec = halt;  // the free ping buffer
  add2_kernel<<<cdiv((long)M_DEC * Dd, 256), 256, 0, stream>>>(cq, fq, hdec, hb, (long)M_DEC * Dd);
  float* hd2 = hcur;   // encoder output no longer needed
  for (int l = 0; l < 3; ++l)
    run_layer(hdec, hd2, M_DEC, S_DEC, NSEQ,
              dwqkvT[l], dwoT[l], dwf1T[l], dwf2T[l],
              dLn1 + (size_t)l * 1024, dLn2 + (size_t)l * 1024,
              dWc + (size_t)l * Cc * Cc);

  // ---- head: tanh(h@Wr1+br1)@Wr2+br2 -> recon directly into d_out ----
  gemm(hb, wr1T, nullptr, t1b, br1, M_DEC, 512, 512, 2);
  {
    dim3 grid(1, M_DEC / 128);
    gemm_n16<<<grid, 256, 0, stream>>>(t1b, wr2T, fo, br2, M_DEC, 512);
  }

  // ---- losses ----
  mse_kernel<<<1024, 256, 0, stream>>>(fo, x, accs + 0, (long)Bb * Cc * Tt);
  finalize_kernel<<<1, 1, 0, stream>>>(accs, fo + 786432);
}
